// SimpleModel_36189394436139
// MI455X (gfx1250) — matrix operand order repaired and checked by reading, not compiled, not run on hardware
//
#include <hip/hip_runtime.h>
#include <cstdint>
#include <cstddef>

// ---------------- CDNA5 WMMA types ----------------
typedef __attribute__((ext_vector_type(16))) __bf16 v16bf;
typedef __attribute__((ext_vector_type(8)))  float  v8f;

#define BATCH 8192
#define KDIM  4096
#define NDIM  4096
#define HMEM  1048576
#define PRIME     2147483647LL
#define PRIME_U   2147483647u
#define HMOD_U    1047552u       // HMEM - 32*32

// Tiling: 128x128 output tile per WG, K-step 32, 8 waves (wave32)
#define BT 128
#define NT 128

// LDS layout (per buffer): 8 n-subtiles * 32 lane-rows * 12 dwords (8 data + 4 pad)
#define LROW   12                // dwords per lane-row (48B, 16B-aligned, 2-way banks)
#define LTILE  (32 * LROW)       // 384 dwords per subtile
#define LBUF   (8 * LTILE)       // 3072 dwords per buffer

__device__ __forceinline__ uint32_t pack_bf16(float lo, float hi) {
#if __has_builtin(__builtin_amdgcn_cvt_pk_bf16_f32)
    auto p = __builtin_amdgcn_cvt_pk_bf16_f32(lo, hi);   // single v_cvt_pk_bf16_f32
    return __builtin_bit_cast(uint32_t, p);
#else
    __bf16 a = (__bf16)lo;
    __bf16 b = (__bf16)hi;
    uint16_t ua = __builtin_bit_cast(uint16_t, a);
    uint16_t ub = __builtin_bit_cast(uint16_t, b);
    return ((uint32_t)ub << 16) | (uint32_t)ua;
#endif
}

__global__ void __launch_bounds__(256)
rz_zero_ws(float* p, int n) {
    int i = blockIdx.x * blockDim.x + threadIdx.x;
    if (i < n) p[i] = 0.0f;
}

__global__ void __launch_bounds__(256)
rz_sqrt_out(const float* __restrict__ in, float* __restrict__ out, int n) {
    int i = blockIdx.x * blockDim.x + threadIdx.x;
    if (i < n) out[i] = sqrtf(in[i]);
}

// Main fused kernel: hashed-gather bf16 GEMM tile + squared-row-sum reduction.
// grid = (64, 32): blockIdx.x -> 128-row tile, blockIdx.y -> 128-col tile.
__global__ void __launch_bounds__(256)
rz_gemm_norm(const float* __restrict__ x,
             const float* __restrict__ hw,
             float* __restrict__ rowsq,
             long long R1, long long R2, unsigned int R3u)
{
    __shared__ __align__(16) uint32_t ldsB[2 * LBUF];    // 24 KB double-buffered

    const int tid  = threadIdx.x;
    const int lane = tid & 31;
    const int wave = tid >> 5;            // 0..7
    const int m    = lane & 15;
    const int h    = lane >> 4;

    const int rowBase = blockIdx.x * BT + wave * 16;
    const int n0      = blockIdx.y * NT;
    const int nb0     = n0 >> 5;

    // producer decode: fixed column per thread, k-pairs split by tid>>7
    const int pn     = tid & 127;
    const int pkbase = tid >> 7;          // 0 or 1
    const int nloc   = (n0 + pn) & 31;

    // Running tile-hash state (wave-uniform: one nb per wave):
    //   s = (kb*R3 + nb*R2 + R1) mod P,  kept reduced in uint32 (P < 2^31).
    uint32_t s32 = (uint32_t)(((long long)(nb0 + (pn >> 5)) * R2 + R1) % PRIME);

    v8f acc[8];
#pragma unroll
    for (int t = 0; t < 8; ++t)
#pragma unroll
        for (int e = 0; e < 8; ++e) acc[t][e] = 0.0f;

    const float* xrow = x + (size_t)(rowBase + m) * KDIM;

    float pf[16];

    // producer: gather 16 coalesced fp32 of column `nloc` of the current hash tile
    auto prod_load = [&]() {
        // wave-uniform tile start -> scalarize so loads use an SGPR base
        int st = __builtin_amdgcn_readfirstlane((int)(s32 % HMOD_U));
        const float* wp = hw + st + nloc;
#pragma unroll
        for (int i = 0; i < 8; ++i) {
            int k = (pkbase + 2 * i) << 1;
            pf[2 * i]     = wp[k * 32];
            pf[2 * i + 1] = wp[(k + 1) * 32];
        }
    };
    // pack bf16 pairs and scatter into WMMA-B fragment order; advance hash state
    auto prod_store = [&](int bufbase) {
#pragma unroll
        for (int i = 0; i < 8; ++i) {
            int k    = (pkbase + 2 * i) << 1;
            int srow = ((k >> 4) << 4) + (pn & 15);      // lane-row = 16*(k>=16) + n%16
            int di   = bufbase + (pn >> 4) * LTILE + srow * LROW + ((k & 15) >> 1);
            ldsB[di] = pack_bf16(pf[2 * i], pf[2 * i + 1]);
        }
        s32 += R3u;                       // < 2^32, stays exact
        if (s32 >= PRIME_U) s32 -= PRIME_U;
    };
    // consumer: A fragment + 8 WMMAs with distance-1 B prefetch
    auto consume = [&](int k0, int curbase) {
        const float* c0 = xrow + k0 + 16 * h;
        const float* c1 = xrow + k0 + 16 * h + 8;
        float4 f0 = *(const float4*)(c0);
        float4 f1 = *(const float4*)(c0 + 4);
        float4 f2 = *(const float4*)(c1);
        float4 f3 = *(const float4*)(c1 + 4);
        v16bf a;
        a[0]=(__bf16)f0.x;  a[1]=(__bf16)f0.y;  a[2]=(__bf16)f0.z;  a[3]=(__bf16)f0.w;
        a[4]=(__bf16)f1.x;  a[5]=(__bf16)f1.y;  a[6]=(__bf16)f1.z;  a[7]=(__bf16)f1.w;
        a[8]=(__bf16)f2.x;  a[9]=(__bf16)f2.y;  a[10]=(__bf16)f2.z; a[11]=(__bf16)f2.w;
        a[12]=(__bf16)f3.x; a[13]=(__bf16)f3.y; a[14]=(__bf16)f3.z; a[15]=(__bf16)f3.w;

        auto loadB = [&](int t) -> v16bf {
            const uint32_t* src = &ldsB[curbase + t * LTILE + lane * LROW];
            union { uint4 q[2]; v16bf v; } u;
            u.q[0] = *(const uint4*)(src);
            u.q[1] = *(const uint4*)(src + 4);
            return u.v;
        };
        v16bf bcur = loadB(0);
        v16bf bnxt;
#pragma unroll
        for (int t = 0; t < 8; ++t) {
            if (t < 7) bnxt = loadB(t + 1);              // keep 2 DS loads in flight
            acc[t] = __builtin_amdgcn_wmma_f32_16x16x32_bf16(
                false, a, false, bcur, (short)0, acc[t], false, false);
            bcur = bnxt;
        }
    };

    // ---- prologue: stage tile kb=0 into buffer 0 ----
    prod_load();
    prod_store(0);
    __syncthreads();

    // ---- steady state: consume tile kb=step, produce tile kb=step+1 ----
    for (int step = 0; step < 127; ++step) {
        const int curbase = (step & 1) ? LBUF : 0;
        prod_load();                       // issue gathers early (overlap WMMA chain)
        consume(step << 5, curbase);
        prod_store(curbase ^ LBUF);
        __syncthreads();
    }
    // last tile (127 & 1 -> buffer 1), nothing to produce
    consume(127 << 5, LBUF);

    // ---- epilogue: per-row sum of squares over this WG's 128 columns ----
#pragma unroll
    for (int r = 0; r < 8; ++r) {
        float sq = 0.0f;
#pragma unroll
        for (int t = 0; t < 8; ++t) {
            float v = acc[t][r];
            sq = fmaf(v, v, sq);
        }
        for (int off = 1; off < 16; off <<= 1)
            sq += __shfl_xor(sq, off, 16);
        if ((lane & 15) == 0) {
            int row = rowBase + (h ? 8 : 0) + r;   // VGPR r -> M=r (lanes 0-15), M=8+r (16-31)
            atomicAdd(&rowsq[row], sq);
        }
    }
}

// ---------------- host: reproduce numpy RandomState(1367).randint(1, 2**31-1, 3) ----------------
namespace {
struct MT19937 {
    uint32_t mt[624];
    int idx;
    void seed(uint32_t s) {
        mt[0] = s;
        for (int i = 1; i < 624; ++i)
            mt[i] = 1812433253u * (mt[i - 1] ^ (mt[i - 1] >> 30)) + (uint32_t)i;
        idx = 624;
    }
    uint32_t next() {
        if (idx >= 624) {
            for (int i = 0; i < 624; ++i) {
                uint32_t y = (mt[i] & 0x80000000u) | (mt[(i + 1) % 624] & 0x7fffffffu);
                uint32_t v = mt[(i + 397) % 624] ^ (y >> 1);
                if (y & 1u) v ^= 2567483615u;
                mt[i] = v;
            }
            idx = 0;
        }
        uint32_t y = mt[idx++];
        y ^= y >> 11;
        y ^= (y << 7) & 2636928640u;
        y ^= (y << 15) & 4022730752u;
        y ^= y >> 18;
        return y;
    }
};
} // namespace

extern "C" void kernel_launch(void* const* d_in, const int* in_sizes, int n_in,
                              void* d_out, int out_size, void* d_ws, size_t ws_size,
                              hipStream_t stream) {
    (void)in_sizes; (void)n_in; (void)out_size; (void)ws_size;
    const float* x  = (const float*)d_in[0];
    const float* hw = (const float*)d_in[1];
    float* out   = (float*)d_out;
    float* rowsq = (float*)d_ws;     // 8192 fp32 accumulators

    // R1,R2,R3: legacy numpy bounded randint; range 2^31-2 fits in 32 bits ->
    // single 32-bit draws, mask 0x7FFFFFFF, reject > 2^31-3, offset 1.
    MT19937 mt;
    mt.seed(1367u);
    long long R[3];
    for (int i = 0; i < 3; ++i) {
        uint32_t v;
        do { v = mt.next() & 0x7fffffffu; } while (v > 0x7ffffffdu);
        R[i] = 1LL + (long long)v;
    }

    rz_zero_ws<<<(BATCH + 255) / 256, 256, 0, stream>>>(rowsq, BATCH);

    dim3 grid(BATCH / BT, NDIM / NT);   // 64 x 32
    rz_gemm_norm<<<grid, 256, 0, stream>>>(x, hw, rowsq, R[0], R[1], (unsigned int)R[2]);

    rz_sqrt_out<<<(BATCH + 255) / 256, 256, 0, stream>>>(rowsq, out, BATCH);
}
